// CrossViewSwapAttention_11819749998928
// MI455X (gfx1250) — matrix operand order repaired and checked by reading, not compiled, not run on hardware
//
#include <hip/hip_runtime.h>
#include <math.h>

// ---------------------------------------------------------------------------
// CrossViewSwapAttention for MI455X (gfx1250, wave32, WMMA).
// All GEMM-shaped work lowered to v_wmma_f32_16x16x32_f16.
// GEMM: 32x64 C tile per wave (2 A-frags x 4 B-frags -> 8 WMMAs / K-step,
// 1.5 b128 loads per WMMA). Attention fused flash-style (online softmax),
// P re-fragmented via per-wave LDS.
// ---------------------------------------------------------------------------

#define NV    6
#define DIMC  128
#define FHH   32
#define FWW   128
#define HHH   64
#define WWW   128
#define HEADS 4
#define DHD   32
#define LWIN  32
#define SQ    1536   // q rows per window  (NV*16*16)
#define SK    768    // kv rows per window (NV*8*16)

typedef __attribute__((ext_vector_type(16))) _Float16 v16h;
typedef __attribute__((ext_vector_type(8)))  _Float16 v8h;
typedef __attribute__((ext_vector_type(8)))  float    v8f;

__device__ __forceinline__ v16h cat8(v8h lo, v8h hi) {
  return __builtin_shufflevector(lo, hi, 0,1,2,3,4,5,6,7,8,9,10,11,12,13,14,15);
}

__device__ __forceinline__ v8f wmma32(v16h a, v16h b, v8f c) {
  return __builtin_amdgcn_wmma_f32_16x16x32_f16(false, a, false, b, (short)0, c, false, false);
}

// A fragment (16x32 f16, row-major, row stride lda halves).
// lane&15 = M row; (lane>>4)*16 = K base, 16 contiguous K per lane.
__device__ __forceinline__ v16h load_afrag(const _Float16* A, int lda, int lane) {
  int m  = lane & 15;
  int kb = (lane >> 4) * 16;
  const _Float16* p = A + (size_t)m * lda;
  v8h lo = *(const v8h*)(p + kb);
  v8h hi = *(const v8h*)(p + kb + 8);
  return cat8(lo, hi);
}

// B fragment (32x16) from Bt stored [N][K] (so each lane reads 16 contiguous halves).
__device__ __forceinline__ v16h load_bfrag(const _Float16* Bt, int ldb, int lane) {
  int n  = lane & 15;
  int kb = (lane >> 4) * 16;
  const _Float16* p = Bt + (size_t)n * ldb + kb;
  v8h lo = *(const v8h*)(p);
  v8h hi = *(const v8h*)(p + 8);
  return cat8(lo, hi);
}

__device__ __forceinline__ float wsum32(float v) {
  #pragma unroll
  for (int m = 1; m < 32; m <<= 1) v += __shfl_xor(v, m, 32);
  return v;
}
__device__ __forceinline__ float hmax16(float v) {
  #pragma unroll
  for (int m = 1; m < 16; m <<= 1) v = fmaxf(v, __shfl_xor(v, m, 32));
  return v;
}
__device__ __forceinline__ float hsum16(float v) {
  #pragma unroll
  for (int m = 1; m < 16; m <<= 1) v += __shfl_xor(v, m, 32);
  return v;
}

// ---------------------------------------------------------------------------
// Small prep kernels
// ---------------------------------------------------------------------------

__global__ void k_cembed(const float* __restrict__ Wcam, const float* __restrict__ Einv,
                         float* __restrict__ cemb) {
  int n = blockIdx.x, d = threadIdx.x;
  float s = 0.f;
  #pragma unroll
  for (int j = 0; j < 4; ++j) s += Wcam[d * 4 + j] * Einv[n * 16 + j * 4 + 3];
  cemb[n * 128 + d] = s;
}

// img_embed (gated, channel-normalized), layout (n,fh,fw,d). One wave per pixel.
__global__ void k_imgembed(const float* __restrict__ Iinv, const float* __restrict__ Einv,
                           const float* __restrict__ Wimg, const float* __restrict__ cemb,
                           const float* __restrict__ gatep, float* __restrict__ imge) {
  int lane = threadIdx.x & 31;
  int w = (blockIdx.x * blockDim.x + threadIdx.x) >> 5;
  if (w >= NV * FHH * FWW) return;
  int n = w / (FHH * FWW);
  int p = w % (FHH * FWW);
  int fh = p >> 7, fw = p & 127;
  float px = fw * (1408.0f / 127.0f);
  float py = fh * (512.0f / 31.0f);
  float c0 = Iinv[n*9+0]*px + Iinv[n*9+1]*py + Iinv[n*9+2];
  float c1 = Iinv[n*9+3]*px + Iinv[n*9+4]*py + Iinv[n*9+5];
  float c2 = Iinv[n*9+6]*px + Iinv[n*9+7]*py + Iinv[n*9+8];
  float dv[4];
  #pragma unroll
  for (int i = 0; i < 4; ++i)
    dv[i] = Einv[n*16+i*4+0]*c0 + Einv[n*16+i*4+1]*c1 + Einv[n*16+i*4+2]*c2 + Einv[n*16+i*4+3];
  float e[4]; float sq = 0.f;
  #pragma unroll
  for (int i = 0; i < 4; ++i) {
    int d = lane * 4 + i;
    float t = Wimg[d*4+0]*dv[0] + Wimg[d*4+1]*dv[1] + Wimg[d*4+2]*dv[2] + Wimg[d*4+3]*dv[3];
    t -= cemb[n * 128 + d];
    e[i] = t; sq += t * t;
  }
  float nrm = sqrtf(wsum32(sq)) + 1e-7f;
  float gate = gatep[0];
  #pragma unroll
  for (int i = 0; i < 4; ++i)
    imge[(size_t)w * 128 + lane * 4 + i] = e[i] / nrm * gate;
}

// BN + ReLU of feature, transposed to rows (n,fh,fw)x(c), f16 for both BN param sets.
__global__ void k_featstage(const float* __restrict__ feat,
                            const float* gp, const float* bp, const float* mp, const float* vp,
                            const float* gl, const float* bl, const float* ml, const float* vl,
                            _Float16* __restrict__ featP, _Float16* __restrict__ featL) {
  int i = blockIdx.x * blockDim.x + threadIdx.x;
  if (i >= NV * FHH * FWW * 128) return;
  int p = i >> 7, c = i & 127;
  int n = p >> 12, q = p & 4095;
  int fh = q >> 7, fw = q & 127;
  float f = feat[(size_t)((n * 128 + c) * FHH + fh) * FWW + fw];
  float a = (f - mp[c]) * rsqrtf(vp[c] + 1e-5f) * gp[c] + bp[c];
  float b = (f - ml[c]) * rsqrtf(vl[c] + 1e-5f) * gl[c] + bl[c];
  featP[i] = (_Float16)fmaxf(a, 0.0f);
  featL[i] = (_Float16)fmaxf(b, 0.0f);
}

__global__ void k_cast16(const float* __restrict__ s, _Float16* __restrict__ d, int n) {
  int i = blockIdx.x * blockDim.x + threadIdx.x;
  if (i < n) d[i] = (_Float16)s[i];
}
// W [K][N] row-major -> Bt [N][K] f16
__global__ void k_castBt(const float* __restrict__ W, _Float16* __restrict__ Bt, int Kd, int Nd) {
  int i = blockIdx.x * blockDim.x + threadIdx.x;
  if (i >= Kd * Nd) return;
  int k = i / Nd, n = i % Nd;
  Bt[(size_t)n * Kd + k] = (_Float16)W[i];
}

// ---------------------------------------------------------------------------
// Generic WMMA GEMM: C[M,N] f32 = A16[M,K] @ Bt16[N,K]^T (+bias[N])
// 32x64 C tile per wave: 2 A-frags x 4 B-frags -> 8 WMMAs per 32-K step.
// Requires M % 32 == 0, N % 64 == 0 (true for all call sites).
// ---------------------------------------------------------------------------
__global__ void k_gemm(const _Float16* __restrict__ A, const _Float16* __restrict__ Bt,
                       const float* __restrict__ bias, float* __restrict__ C,
                       int M, int N, int K, int hasBias) {
  int lane = threadIdx.x & 31;
  int wid  = threadIdx.x >> 5;
  int ntiles = N >> 6;
  int tiles  = (M >> 5) * ntiles;
  int t = blockIdx.x * 8 + wid;
  if (t >= tiles) return;
  int tm = (t / ntiles) << 5;
  int tn = (t % ntiles) << 6;
  v8f acc[2][4];
  #pragma unroll
  for (int j = 0; j < 4; ++j) {
    float bv = hasBias ? bias[tn + j * 16 + (lane & 15)] : 0.0f;
    #pragma unroll
    for (int r = 0; r < 8; ++r) { acc[0][j][r] = bv; acc[1][j][r] = bv; }
  }
  for (int kc = 0; kc < K; kc += 32) {
    v16h a0 = load_afrag(A + (size_t)tm * K + kc, K, lane);
    v16h a1 = load_afrag(A + (size_t)(tm + 16) * K + kc, K, lane);
    #pragma unroll
    for (int j = 0; j < 4; ++j) {
      v16h b = load_bfrag(Bt + (size_t)(tn + j * 16) * K + kc, K, lane);
      acc[0][j] = wmma32(a0, b, acc[0][j]);
      acc[1][j] = wmma32(a1, b, acc[1][j]);
    }
  }
  int col = lane & 15;
  int rb  = (lane < 16) ? 0 : 8;
  #pragma unroll
  for (int i = 0; i < 2; ++i)
    #pragma unroll
    for (int j = 0; j < 4; ++j)
      #pragma unroll
      for (int r = 0; r < 8; ++r)
        C[(size_t)(tm + i * 16 + rb + r) * N + (tn + j * 16 + col)] = acc[i][j][r];
}

// ---------------------------------------------------------------------------
// key/val width-LayerNorm (normalizes across FW, gamma/beta indexed by fw).
// One wave per (n, d, fh). Optionally adds gated img_embed first.
// ---------------------------------------------------------------------------
__global__ void k_widthln(const float* __restrict__ conv, const float* __restrict__ img,
                          int useImg, const float* __restrict__ g, const float* __restrict__ b,
                          float* __restrict__ out) {
  int lane = threadIdx.x & 31;
  int w = (blockIdx.x * blockDim.x + threadIdx.x) >> 5;
  if (w >= NV * 128 * FHH) return;
  int n = w / (128 * FHH);
  int rem = w % (128 * FHH);
  int d = rem / FHH;
  int fh = rem % FHH;
  size_t base = ((size_t)n * FHH + fh) * FWW * 128 + d;
  float v[4]; float s = 0.f;
  #pragma unroll
  for (int i = 0; i < 4; ++i) {
    int fw = lane + i * 32;
    float t = conv[base + (size_t)fw * 128];
    if (useImg) t += img[base + (size_t)fw * 128];
    v[i] = t; s += t;
  }
  float mean = wsum32(s) * (1.0f / 128.0f);
  float vs = 0.f;
  #pragma unroll
  for (int i = 0; i < 4; ++i) { float dd = v[i] - mean; vs += dd * dd; }
  float rstd = rsqrtf(wsum32(vs) * (1.0f / 128.0f) + 1e-5f);
  #pragma unroll
  for (int i = 0; i < 4; ++i) {
    int fw = lane + i * 32;
    out[base + (size_t)fw * 128] = (v[i] - mean) * rstd * g[fw] + b[fw];
  }
}

// K/V staging: channel-LN then window permute to rows (l*SK + r). mode0 = round1
// windows, mode1 = round2 "swapped" windows. One wave per source location.
__global__ void k_kvstage(const float* __restrict__ src, const float* __restrict__ g,
                          const float* __restrict__ b, _Float16* __restrict__ dst, int mode) {
  int lane = threadIdx.x & 31;
  int w = (blockIdx.x * blockDim.x + threadIdx.x) >> 5;
  if (w >= NV * FHH * FWW) return;
  int n = w / (FHH * FWW);
  int q = w % (FHH * FWW);
  int fh = q >> 7, fw = q & 127;
  int l, r;
  if (mode == 0) { l = (fh >> 3) * 8 + (fw >> 4); r = n * 128 + (fh & 7) * 16 + (fw & 15); }
  else           { l = (fh & 3) * 8 + (fw & 7);   r = n * 128 + (fh >> 2) * 16 + (fw >> 3); }
  float4 a4 = ((const float4*)(src + (size_t)w * 128))[lane];
  float vv[4] = {a4.x, a4.y, a4.z, a4.w};
  float mean = wsum32(vv[0] + vv[1] + vv[2] + vv[3]) * (1.0f / 128.0f);
  float vs = 0.f;
  #pragma unroll
  for (int i = 0; i < 4; ++i) { float dd = vv[i] - mean; vs += dd * dd; }
  float rstd = rsqrtf(wsum32(vs) * (1.0f / 128.0f) + 1e-5f);
  size_t o = (size_t)(l * SK + r) * 128 + lane * 4;
  #pragma unroll
  for (int i = 0; i < 4; ++i) {
    int d = lane * 4 + i;
    dst[o + i] = (_Float16)((vv[i] - mean) * rstd * g[d] + b[d]);
  }
}

// Round-1 query staging: BEV embed + x, qln LN, window permute. Wave per (n,hh,ww).
__global__ void k_querystage(const float* __restrict__ grid0, const float* __restrict__ xin,
                             const float* __restrict__ Wbev, const float* __restrict__ bbev,
                             const float* __restrict__ cemb, const float* __restrict__ g,
                             const float* __restrict__ b, _Float16* __restrict__ dst) {
  int lane = threadIdx.x & 31;
  int w = (blockIdx.x * blockDim.x + threadIdx.x) >> 5;
  if (w >= NV * HHH * WWW) return;
  int n = w / (HHH * WWW);
  int s = w % (HHH * WWW);
  int hh = s >> 7, ww = s & 127;
  float g0 = grid0[s], g1 = grid0[HHH * WWW + s];
  float e[4]; float sq = 0.f;
  #pragma unroll
  for (int i = 0; i < 4; ++i) {
    int d = lane * 4 + i;
    float t = Wbev[d * 2 + 0] * g0 + Wbev[d * 2 + 1] * g1 + bbev[d] - cemb[n * 128 + d];
    e[i] = t; sq += t * t;
  }
  float nrm = sqrtf(wsum32(sq)) + 1e-7f;
  float qv[4]; float ssum = 0.f;
  #pragma unroll
  for (int i = 0; i < 4; ++i) {
    int d = lane * 4 + i;
    qv[i] = e[i] / nrm + xin[(size_t)d * HHH * WWW + s];
    ssum += qv[i];
  }
  float mean = wsum32(ssum) * (1.0f / 128.0f);
  float vs = 0.f;
  #pragma unroll
  for (int i = 0; i < 4; ++i) { float dd = qv[i] - mean; vs += dd * dd; }
  float rstd = rsqrtf(wsum32(vs) * (1.0f / 128.0f) + 1e-5f);
  int xx = hh >> 4, w1 = hh & 15, yy = ww >> 4, w2 = ww & 15;
  int l = xx * 8 + yy;
  size_t o = (size_t)(l * SQ + n * 256 + w1 * 16 + w2) * 128 + lane * 4;
  #pragma unroll
  for (int i = 0; i < 4; ++i) {
    int d = lane * 4 + i;
    dst[o + i] = (_Float16)((qv[i] - mean) * rstd * g[d] + b[d]);
  }
}

// Round-2 query staging: LN(x_skip) replicated to NV views. Wave per (hh,ww).
__global__ void k_q2stage(const float* __restrict__ xskip, const float* __restrict__ g,
                          const float* __restrict__ b, _Float16* __restrict__ dst) {
  int lane = threadIdx.x & 31;
  int s = (blockIdx.x * blockDim.x + threadIdx.x) >> 5;
  if (s >= HHH * WWW) return;
  int hh = s >> 7, ww = s & 127;
  float4 a4 = ((const float4*)(xskip + (size_t)s * 128))[lane];
  float vv[4] = {a4.x, a4.y, a4.z, a4.w};
  float mean = wsum32(vv[0] + vv[1] + vv[2] + vv[3]) * (1.0f / 128.0f);
  float vs = 0.f;
  #pragma unroll
  for (int i = 0; i < 4; ++i) { float dd = vv[i] - mean; vs += dd * dd; }
  float rstd = rsqrtf(wsum32(vs) * (1.0f / 128.0f) + 1e-5f);
  _Float16 o16[4];
  #pragma unroll
  for (int i = 0; i < 4; ++i) {
    int d = lane * 4 + i;
    o16[i] = (_Float16)((vv[i] - mean) * rstd * g[d] + b[d]);
  }
  int xx = hh >> 4, w1 = hh & 15, yy = ww >> 4, w2 = ww & 15;
  int l = xx * 8 + yy;
  for (int n = 0; n < NV; ++n) {
    size_t o = (size_t)(l * SQ + n * 256 + w1 * 16 + w2) * 128 + lane * 4;
    #pragma unroll
    for (int i = 0; i < 4; ++i) dst[o + i] = o16[i];
  }
}

// Head split (+q scale by (dh^-0.5)/tau). transposed=1 writes [lh][dh][r] (for V).
__global__ void k_headsplit(const float* __restrict__ X, _Float16* __restrict__ out,
                            int S, int transposed, const int* __restrict__ oc, int applyScale) {
  int i = blockIdx.x * blockDim.x + threadIdx.x;
  if (i >= LWIN * S * DIMC) return;
  int row = i >> 7, c = i & 127;
  int h = c >> 5, dh = c & 31;
  int l = row / S, r = row % S;
  float sc = 1.0f;
  if (applyScale) {
    float o = (float)(oc[0] > 0 ? oc[0] : 0);
    float tau = 2.0f / (5.0f + o) + 0.6f;
    tau = fminf(fmaxf(tau, 0.4f), 1.5f);
    sc = 0.17677669529663687f / tau;  // 32^-0.5 / tau
  }
  float v = X[i] * sc;
  size_t o = transposed ? ((size_t)((l * HEADS + h) * DHD + dh) * S + r)
                        : ((size_t)((l * HEADS + h) * S + r) * DHD + dh);
  out[o] = (_Float16)v;
}

// ---------------------------------------------------------------------------
// Fused flash attention. Block = (window l, head h, 128-row q tile); 8 waves,
// each owns 16 q rows. Per 32-key chunk: 2 WMMAs for S, online softmax via
// 16-lane shfl reductions, P staged through per-wave LDS to re-fragment as A,
// then 2 WMMAs accumulate O = P @ V.
// ---------------------------------------------------------------------------
__global__ void k_attn(const _Float16* __restrict__ qh, const _Float16* __restrict__ kh,
                       const _Float16* __restrict__ vhT, _Float16* __restrict__ aout) {
  __shared__ _Float16 Psc[8 * 16 * 32];
  int lane = threadIdx.x & 31;
  int wid  = threadIdx.x >> 5;
  int bid = blockIdx.x;
  int l   = bid / (HEADS * 12);
  int rem = bid % (HEADS * 12);
  int h   = rem / 12;
  int qt  = rem % 12;
  int lh  = l * HEADS + h;
  int qr0 = qt * 128 + wid * 16;

  const _Float16* Qb = qh  + (size_t)(lh * SQ + qr0) * DHD;
  const _Float16* Kb = kh  + (size_t)lh * SK * DHD;
  const _Float16* Vb = vhT + (size_t)lh * DHD * SK;

  v16h qf = load_afrag(Qb, DHD, lane);

  float rmax[8], rsum[8];
  v8f O0 = {0.f,0.f,0.f,0.f,0.f,0.f,0.f,0.f};
  v8f O1 = {0.f,0.f,0.f,0.f,0.f,0.f,0.f,0.f};
  #pragma unroll
  for (int r = 0; r < 8; ++r) { rmax[r] = -1e30f; rsum[r] = 0.0f; }

  _Float16* Pw = Psc + wid * (16 * 32);
  int bn  = lane & 15;
  int bkb = (lane >> 4) * 16;

  for (int kc = 0; kc < SK; kc += 32) {
    const _Float16* pk0 = Kb + (size_t)(kc + bn) * DHD + bkb;
    const _Float16* pk1 = Kb + (size_t)(kc + 16 + bn) * DHD + bkb;
    v16h b0 = cat8(*(const v8h*)pk0, *(const v8h*)(pk0 + 8));
    v16h b1 = cat8(*(const v8h*)pk1, *(const v8h*)(pk1 + 8));
    v8f Z = {0.f,0.f,0.f,0.f,0.f,0.f,0.f,0.f};
    v8f S0 = wmma32(qf, b0, Z);
    v8f S1 = wmma32(qf, b1, Z);
    #pragma unroll
    for (int r = 0; r < 8; ++r) {
      float cm = hmax16(fmaxf(S0[r], S1[r]));
      float nm = fmaxf(rmax[r], cm);
      float corr = __expf(rmax[r] - nm);
      rmax[r] = nm;
      float p0 = __expf(S0[r] - nm);
      float p1 = __expf(S1[r] - nm);
      float ps = hsum16(p0 + p1);
      rsum[r] = rsum[r] * corr + ps;
      O0[r] *= corr; O1[r] *= corr;
      int row = (lane < 16) ? r : (r + 8);
      Pw[row * 32 + bn]      = (_Float16)p0;
      Pw[row * 32 + 16 + bn] = (_Float16)p1;
    }
    v16h pf = load_afrag(Pw, 32, lane);
    const _Float16* pv0 = Vb + (size_t)(bn) * SK + kc + bkb;
    const _Float16* pv1 = Vb + (size_t)(16 + bn) * SK + kc + bkb;
    v16h vb0 = cat8(*(const v8h*)pv0, *(const v8h*)(pv0 + 8));
    v16h vb1 = cat8(*(const v8h*)pv1, *(const v8h*)(pv1 + 8));
    O0 = wmma32(pf, vb0, O0);
    O1 = wmma32(pf, vb1, O1);
  }
  #pragma unroll
  for (int r = 0; r < 8; ++r) {
    int row = (lane < 16) ? r : (r + 8);
    float inv = 1.0f / rsum[r];
    size_t o = (size_t)(l * SQ + qr0 + row) * DIMC + h * DHD;
    aout[o + bn]      = (_Float16)(O0[r] * inv);
    aout[o + 16 + bn] = (_Float16)(O1[r] * inv);
  }
}

// mean over views + skip, un-window to (hh,ww,d)
__global__ void k_zmerge(const float* __restrict__ Z, const float* __restrict__ skip,
                         int chanFirst, float* __restrict__ out) {
  int i = blockIdx.x * blockDim.x + threadIdx.x;
  if (i >= HHH * WWW * 128) return;
  int s = i >> 7, d = i & 127;
  int hh = s >> 7, ww = s & 127;
  int xx = hh >> 4, w1 = hh & 15, yy = ww >> 4, w2 = ww & 15;
  int l = xx * 8 + yy;
  size_t base = (size_t)(l * SQ + w1 * 16 + w2) * 128 + d;
  float sum = 0.f;
  #pragma unroll
  for (int n = 0; n < NV; ++n) sum += Z[base + (size_t)n * 256 * 128];
  float sk = chanFirst ? skip[(size_t)d * HHH * WWW + s] : skip[i];
  out[i] = sum * (1.0f / 6.0f) + sk;
}

__global__ void k_lncast(const float* __restrict__ src, const float* __restrict__ g,
                         const float* __restrict__ b, _Float16* __restrict__ dst, int rows) {
  int lane = threadIdx.x & 31;
  int s = (blockIdx.x * blockDim.x + threadIdx.x) >> 5;
  if (s >= rows) return;
  float4 a4 = ((const float4*)(src + (size_t)s * 128))[lane];
  float vv[4] = {a4.x, a4.y, a4.z, a4.w};
  float mean = wsum32(vv[0] + vv[1] + vv[2] + vv[3]) * (1.0f / 128.0f);
  float vs = 0.f;
  #pragma unroll
  for (int i = 0; i < 4; ++i) { float dd = vv[i] - mean; vs += dd * dd; }
  float rstd = rsqrtf(wsum32(vs) * (1.0f / 128.0f) + 1e-5f);
  #pragma unroll
  for (int i = 0; i < 4; ++i) {
    int d = lane * 4 + i;
    dst[(size_t)s * 128 + d] = (_Float16)((vv[i] - mean) * rstd * g[d] + b[d]);
  }
}

__global__ void k_gelu(const float* __restrict__ src, _Float16* __restrict__ dst, int n) {
  int i = blockIdx.x * blockDim.x + threadIdx.x;
  if (i < n) {
    float v = src[i];
    dst[i] = (_Float16)(0.5f * v * (1.0f + erff(v * 0.7071067811865476f)));
  }
}

__global__ void k_add(const float* __restrict__ a, const float* __restrict__ b,
                      float* __restrict__ o, int n) {
  int i = blockIdx.x * blockDim.x + threadIdx.x;
  if (i < n) o[i] = a[i] + b[i];
}

// post-LN + transpose to (d, hh, ww)
__global__ void k_final(const float* __restrict__ src, const float* __restrict__ g,
                        const float* __restrict__ b, float* __restrict__ y) {
  int lane = threadIdx.x & 31;
  int s = (blockIdx.x * blockDim.x + threadIdx.x) >> 5;
  if (s >= HHH * WWW) return;
  float4 a4 = ((const float4*)(src + (size_t)s * 128))[lane];
  float vv[4] = {a4.x, a4.y, a4.z, a4.w};
  float mean = wsum32(vv[0] + vv[1] + vv[2] + vv[3]) * (1.0f / 128.0f);
  float vs = 0.f;
  #pragma unroll
  for (int i = 0; i < 4; ++i) { float dd = vv[i] - mean; vs += dd * dd; }
  float rstd = rsqrtf(wsum32(vs) * (1.0f / 128.0f) + 1e-5f);
  #pragma unroll
  for (int i = 0; i < 4; ++i) {
    int d = lane * 4 + i;
    y[(size_t)d * HHH * WWW + s] = (vv[i] - mean) * rstd * g[d] + b[d];
  }
}

// ---------------------------------------------------------------------------
// Host orchestration.
// Input order = setup_inputs() insertion order; params flattened in insertion
// order (attp() sub-dicts and mlp dicts likewise).
// ---------------------------------------------------------------------------
extern "C" void kernel_launch(void* const* d_in, const int* in_sizes, int n_in,
                              void* d_out, int out_size, void* d_ws, size_t ws_size,
                              hipStream_t stream) {
  (void)in_sizes; (void)n_in; (void)out_size; (void)ws_size;
  auto F = [&](int i) { return (const float*)d_in[i]; };
  const float* X    = F(1);
  const float* G0   = F(2);
  const float* FEAT = F(3);
  const float* IINV = F(4);
  const float* EINV = F(5);
  const int*   OCNT = (const int*)d_in[6];
  const float *bn_fl_g = F(7),  *bn_fl_b = F(8),  *bn_fl_m = F(9),  *bn_fl_v = F(10);
  const float *bn_fp_g = F(11), *bn_fp_b = F(12), *bn_fp_m = F(13), *bn_fp_v = F(14);
  const float *W_fl = F(15), *W_fp = F(16), *W_bev = F(17), *b_bev = F(18);
  const float *W_img = F(19), *W_cam = F(20), *gate = F(21);
  const float *key_g = F(22), *key_b = F(23), *val_g = F(24), *val_b = F(25);
  const int A1 = 26, A2 = 40;  // qln_g,qln_b,kln_g,kln_b,vln_g,vln_b,Wq,bq,Wk,bk,Wv,bv,Wp,bp
  const float *pre1_g = F(54), *pre1_b = F(55), *pre2_g = F(56), *pre2_b = F(57);
  const float *post_g = F(58), *post_b = F(59);
  const int M1 = 60, M2 = 64;  // W1,b1,W2,b2

  size_t off = 0;
  auto af = [&](size_t n) { float* p = (float*)((char*)d_ws + off);
                            off += (n * sizeof(float) + 255) & ~(size_t)255; return p; };
  auto ah = [&](size_t n) { _Float16* p = (_Float16*)((char*)d_ws + off);
                            off += (n * sizeof(_Float16) + 255) & ~(size_t)255; return p; };

  float* cemb = af(768);
  float* Qf   = af(6291456);   // aliased: keyconv (pre-LN key)
  float* Kf   = af(3145728);   // aliased: valconv
  float* Vf   = af(3145728);   // aliased: imge
  float* kLN  = af(3145728);
  float* vLN  = af(3145728);
  float* Zf   = af(6291456);
  float* q2   = af(1048576);
  float* Hbuf = af(2097152);
  float* mlpo = af(1048576);
  float* xskip= af(1048576);
  float* q3   = af(1048576);
  float* q3b  = af(1048576);
  _Float16* qstage = ah(6291456);  // aliased: featP (first half) / featL (second half)
  _Float16* kstage = ah(3145728);
  _Float16* vstage = ah(3145728);
  _Float16* qhB = ah(6291456);
  _Float16* khB = ah(3145728);
  _Float16* vhT = ah(3145728);
  _Float16* aoB = ah(6291456);
  _Float16* ln16 = ah(1048576);
  _Float16* Hg   = ah(2097152);
  _Float16* WfpBt = ah(16384); _Float16* WflBt = ah(16384);
  _Float16* Wq1t = ah(16384); _Float16* Wk1t = ah(16384);
  _Float16* Wv1t = ah(16384); _Float16* Wp1t = ah(16384);
  _Float16* Wq2t = ah(16384); _Float16* Wk2t = ah(16384);
  _Float16* Wv2t = ah(16384); _Float16* Wp2t = ah(16384);
  _Float16* W1a = ah(32768); _Float16* W2a = ah(32768);
  _Float16* W1b = ah(32768); _Float16* W2b = ah(32768);

  float* keyconv = Qf; float* valconv = Kf; float* imge = Vf;
  _Float16* featP = qstage; _Float16* featL = qstage + 3145728;

  auto gemm = [&](const _Float16* A, const _Float16* Bt, const float* bias, float* C,
                  int M, int Nn, int Kk) {
    int tiles = (M / 32) * (Nn / 64);
    k_gemm<<<(tiles + 7) / 8, 256, 0, stream>>>(A, Bt, bias, C, M, Nn, Kk, bias ? 1 : 0);
  };

  // --- embeddings & feature prep ---
  k_cembed<<<6, 128, 0, stream>>>(W_cam, EINV, cemb);
  k_imgembed<<<3072, 256, 0, stream>>>(IINV, EINV, W_img, cemb, gate, imge);
  k_featstage<<<12288, 256, 0, stream>>>(FEAT, bn_fp_g, bn_fp_b, bn_fp_m, bn_fp_v,
                                         bn_fl_g, bn_fl_b, bn_fl_m, bn_fl_v, featP, featL);
  // --- weight staging ---
  k_cast16<<<64, 256, 0, stream>>>(W_fp, WfpBt, 16384);   // W_fp is already [out][in] = Bt
  k_cast16<<<64, 256, 0, stream>>>(W_fl, WflBt, 16384);
  k_castBt<<<64, 256, 0, stream>>>(F(A1 + 6),  Wq1t, 128, 128);
  k_castBt<<<64, 256, 0, stream>>>(F(A1 + 8),  Wk1t, 128, 128);
  k_castBt<<<64, 256, 0, stream>>>(F(A1 + 10), Wv1t, 128, 128);
  k_castBt<<<64, 256, 0, stream>>>(F(A1 + 12), Wp1t, 128, 128);
  k_castBt<<<64, 256, 0, stream>>>(F(A2 + 6),  Wq2t, 128, 128);
  k_castBt<<<64, 256, 0, stream>>>(F(A2 + 8),  Wk2t, 128, 128);
  k_castBt<<<64, 256, 0, stream>>>(F(A2 + 10), Wv2t, 128, 128);
  k_castBt<<<64, 256, 0, stream>>>(F(A2 + 12), Wp2t, 128, 128);
  k_castBt<<<128, 256, 0, stream>>>(F(M1 + 0), W1a, 128, 256);
  k_castBt<<<128, 256, 0, stream>>>(F(M1 + 2), W2a, 256, 128);
  k_castBt<<<128, 256, 0, stream>>>(F(M2 + 0), W1b, 128, 256);
  k_castBt<<<128, 256, 0, stream>>>(F(M2 + 2), W2b, 256, 128);

  // --- 1x1 convs (WMMA GEMMs) + width-LN ---
  gemm(featP, WfpBt, nullptr, keyconv, 24576, 128, 128);
  gemm(featL, WflBt, nullptr, valconv, 24576, 128, 128);
  k_widthln<<<3072, 256, 0, stream>>>(keyconv, imge, 1, key_g, key_b, kLN);
  k_widthln<<<3072, 256, 0, stream>>>(valconv, nullptr, 0, val_g, val_b, vLN);

  auto attnRound = [&](int AP, _Float16* Wqt, _Float16* Wkt, _Float16* Wvt, _Float16* Wpt) {
    gemm(qstage, Wqt, F(AP + 7),  Qf, 49152, 128, 128);
    gemm(kstage, Wkt, F(AP + 9),  Kf, 24576, 128, 128);
    gemm(vstage, Wvt, F(AP + 11), Vf, 24576, 128, 128);
    k_headsplit<<<24576, 256, 0, stream>>>(Qf, qhB, SQ, 0, OCNT, 1);
    k_headsplit<<<12288, 256, 0, stream>>>(Kf, khB, SK, 0, OCNT, 0);
    k_headsplit<<<12288, 256, 0, stream>>>(Vf, vhT, SK, 1, OCNT, 0);
    k_attn<<<1536, 256, 0, stream>>>(qhB, khB, vhT, aoB);
    gemm(aoB, Wpt, F(AP + 13), Zf, 49152, 128, 128);
  };

  // --- round 1 ---
  k_querystage<<<6144, 256, 0, stream>>>(G0, X, W_bev, b_bev, cemb, F(A1 + 0), F(A1 + 1), qstage);
  k_kvstage<<<3072, 256, 0, stream>>>(kLN, F(A1 + 2), F(A1 + 3), kstage, 0);
  k_kvstage<<<3072, 256, 0, stream>>>(vLN, F(A1 + 4), F(A1 + 5), vstage, 0);
  attnRound(A1, Wq1t, Wk1t, Wv1t, Wp1t);
  k_zmerge<<<4096, 256, 0, stream>>>(Zf, X, 1, q2);
  k_lncast<<<1024, 256, 0, stream>>>(q2, pre1_g, pre1_b, ln16, 8192);
  gemm(ln16, W1a, F(M1 + 1), Hbuf, 8192, 256, 128);
  k_gelu<<<8192, 256, 0, stream>>>(Hbuf, Hg, 2097152);
  gemm(Hg, W2a, F(M1 + 3), mlpo, 8192, 128, 256);
  k_add<<<4096, 256, 0, stream>>>(q2, mlpo, xskip, 1048576);

  // --- round 2 (swapped windows) ---
  k_q2stage<<<1024, 256, 0, stream>>>(xskip, F(A2 + 0), F(A2 + 1), qstage);
  k_kvstage<<<3072, 256, 0, stream>>>(kLN, F(A2 + 2), F(A2 + 3), kstage, 1);
  k_kvstage<<<3072, 256, 0, stream>>>(vLN, F(A2 + 4), F(A2 + 5), vstage, 1);
  attnRound(A2, Wq2t, Wk2t, Wv2t, Wp2t);
  k_zmerge<<<4096, 256, 0, stream>>>(Zf, xskip, 0, q3);
  k_lncast<<<1024, 256, 0, stream>>>(q3, pre2_g, pre2_b, ln16, 8192);
  gemm(ln16, W1b, F(M2 + 1), Hbuf, 8192, 256, 128);
  k_gelu<<<8192, 256, 0, stream>>>(Hbuf, Hg, 2097152);
  gemm(Hg, W2b, F(M2 + 3), mlpo, 8192, 128, 256);
  k_add<<<4096, 256, 0, stream>>>(q3, mlpo, q3b, 1048576);
  k_final<<<1024, 256, 0, stream>>>(q3b, post_g, post_b, (float*)d_out);
}
